// GAT_11828339933316
// MI455X (gfx1250) — compile-verified, matrix operand order checked by reading, not run on hardware
//
#include <hip/hip_runtime.h>
#include <hip/hip_bf16.h>

// ---------------- problem constants (match reference) ----------------
#define NN    50000      // nodes
#define EDGES 800000     // edges before self loops
#define ETOT  850000     // edges + self loops
#define HH1   8          // heads layer 1
#define CC    64         // channels per head
#define HC    512        // H*C
#define FIN   128        // input features
#define NG    128        // graphs
#define NCLS  10         // classes
#define NEGS  0.2f

typedef _Float16 half_t;
typedef __attribute__((ext_vector_type(16))) _Float16 v16h;
typedef __attribute__((ext_vector_type(8)))  float    v8f;

// ---------------- helpers ----------------
__device__ __forceinline__ void edge_sd(const int* __restrict__ ei, int e, int& s, int& d) {
    if (e < EDGES) { s = ei[e]; d = ei[EDGES + e]; }
    else           { s = e - EDGES; d = s; }
}

__device__ __forceinline__ void atomicMaxF(float* addr, float val) {
    int old = __float_as_int(*addr);
    while (__int_as_float(old) < val) {
        int prev = atomicCAS((int*)addr, old, __float_as_int(val));
        if (prev == old) break;
        old = prev;
    }
}

// ---------------- fills / conversions ----------------
__global__ void gat_fill(float* __restrict__ p, long n, float v) {
    long i = (long)blockIdx.x * blockDim.x + threadIdx.x;
    if (i < n) p[i] = v;
}

__global__ void gat_f32_to_f16(const float* __restrict__ in, half_t* __restrict__ out, long n) {
    long i = (long)blockIdx.x * blockDim.x + threadIdx.x;
    if (i < n) out[i] = (half_t)in[i];
}

// ---------------- pack weight matrix B[K x Ncol] (f32 row-major) into WMMA B-fragment order:
// frag index (g, kc, j, lane) -> 16 contiguous halves; g = 64-col group, kc = K/32 chunk, j = 16-col subtile
__global__ void gat_pack_b(const float* __restrict__ B, half_t* __restrict__ Bp, int K, int Ncol) {
    int idx = blockIdx.x * blockDim.x + threadIdx.x;
    int kchunks = K >> 5;
    int total = (Ncol >> 6) * kchunks * 4 * 32;
    if (idx >= total) return;
    int lane = idx & 31;
    int j    = (idx >> 5) & 3;
    int rest = idx >> 7;
    int kc   = rest % kchunks;
    int g    = rest / kchunks;
    int koff = (lane >> 4) << 3;                // 0 or 8
    int n    = (g << 6) + (j << 4) + (lane & 15);
    int kb   = kc << 5;
    half_t* dst = Bp + ((long)idx << 4);
#pragma unroll
    for (int i = 0; i < 8; ++i) {
        dst[i]     = (half_t)B[(long)(kb + koff + i) * Ncol + n];
        dst[i + 8] = (half_t)B[(long)(kb + 16 + koff + i) * Ncol + n];
    }
}

// ---------------- WMMA GEMM: Ch[M x Ncol] (f16) = A[M x K] (f16) @ Bp (packed f16), f32 acc
// one wave computes a 16x64 block (4 accumulators); K mult of 32, M mult of 16, Ncol mult of 64
__global__ __launch_bounds__(256) void gat_wmma_gemm(
    const half_t* __restrict__ A, const half_t* __restrict__ Bp, half_t* __restrict__ Ch,
    int M, int K, int Ncol)
{
    int wid  = threadIdx.x >> 5;
    int lane = threadIdx.x & 31;
    int groups_n = Ncol >> 6;
    int kchunks  = K >> 5;
    int total    = (M >> 4) * groups_n;
    int wg = blockIdx.x * 8 + wid;
    if (wg >= total) return;                 // wave-uniform exit: EXEC stays all-1s
    int tm = wg / groups_n;
    int tg = wg - tm * groups_n;

    int m    = (tm << 4) + (lane & 15);      // A row for this lane
    int koff = (lane >> 4) << 3;             // 0 or 8
    const half_t* arow = A + (long)m * K + koff;

    v8f acc[4] = {};
    for (int kc = 0; kc < kchunks; ++kc) {
        v16h a;
        const half_t* ap = arow + (kc << 5);
#pragma unroll
        for (int i = 0; i < 8; ++i) { a[i] = ap[i]; a[i + 8] = ap[16 + i]; }
#pragma unroll
        for (int j = 0; j < 4; ++j) {
            const half_t* bp = Bp + ((((long)tg * kchunks + kc) * 4 + j) * 32 + lane) * 16;
            v16h b;
#pragma unroll
            for (int i = 0; i < 16; ++i) b[i] = bp[i];
            acc[j] = __builtin_amdgcn_wmma_f32_16x16x32_f16(false, a, false, b,
                                                            (short)0, acc[j], false, false);
        }
    }
    int row0 = (tm << 4) + ((lane >> 4) << 3);
#pragma unroll
    for (int j = 0; j < 4; ++j) {
        int col = (tg << 6) + (j << 4) + (lane & 15);
#pragma unroll
        for (int r = 0; r < 8; ++r)
            Ch[(long)(row0 + r) * Ncol + col] = (half_t)acc[j][r];
    }
}

// ---------------- per-node attention logits from f16 features ----------------
__global__ void gat_attn_logits(const half_t* __restrict__ h,
                                const float* __restrict__ att_s,
                                const float* __restrict__ att_d,
                                float* __restrict__ as_, float* __restrict__ ad_,
                                int n_nodes, int heads, int ch)
{
    int wid  = threadIdx.x >> 5;
    int lane = threadIdx.x & 31;
    int node = blockIdx.x * (blockDim.x >> 5) + wid;
    if (node >= n_nodes) return;
    for (int hh = 0; hh < heads; ++hh) {
        const half_t* base = h + (long)node * heads * ch + (long)hh * ch;
        float ps = 0.f, pd = 0.f;
        for (int c = lane; c < ch; c += 32) {
            float v = (float)base[c];
            ps += v * att_s[hh * ch + c];
            pd += v * att_d[hh * ch + c];
        }
        for (int off = 16; off; off >>= 1) {
            ps += __shfl_xor(ps, off, 32);
            pd += __shfl_xor(pd, off, 32);
        }
        if (lane == 0) { as_[node * heads + hh] = ps; ad_[node * heads + hh] = pd; }
    }
}

// ---------------- segment-softmax pass 1: running max per (dst, head) ----------------
__global__ void gat_edge_max(const int* __restrict__ ei,
                             const float* __restrict__ as_, const float* __restrict__ ad_,
                             float* __restrict__ emax, int heads)
{
    long i = (long)blockIdx.x * blockDim.x + threadIdx.x;
    if (i >= (long)ETOT * heads) return;
    int e  = (int)(i / heads);
    int hh = (int)(i - (long)e * heads);
    int s, d; edge_sd(ei, e, s, d);
    float v = as_[s * heads + hh] + ad_[d * heads + hh];
    v = v > 0.f ? v : NEGS * v;
    atomicMaxF(&emax[d * heads + hh], v);
}

// ---------------- pass 2: ex = exp(e - max), denom += ex ----------------
__global__ void gat_edge_exp(const int* __restrict__ ei,
                             const float* __restrict__ as_, const float* __restrict__ ad_,
                             const float* __restrict__ emax,
                             float* __restrict__ ex, float* __restrict__ den, int heads)
{
    long i = (long)blockIdx.x * blockDim.x + threadIdx.x;
    if (i >= (long)ETOT * heads) return;
    int e  = (int)(i / heads);
    int hh = (int)(i - (long)e * heads);
    int s, d; edge_sd(ei, e, s, d);
    float v = as_[s * heads + hh] + ad_[d * heads + hh];
    v = v > 0.f ? v : NEGS * v;
    float xv = expf(v - emax[d * heads + hh]);
    ex[(long)e * heads + hh] = xv;
    atomicAdd(&den[d * heads + hh], xv);
}

// ---------------- pass 3: out[dst] += h[src] * alpha ; one block per edge, thread per channel
__global__ void gat_edge_aggregate(const int* __restrict__ ei,
                                   const half_t* __restrict__ feat,
                                   const float* __restrict__ ex, const float* __restrict__ den,
                                   float* __restrict__ out, float* __restrict__ alpha_out,
                                   int heads, int ctot)
{
    int e = blockIdx.x;
    int t = threadIdx.x;
    int s, d; edge_sd(ei, e, s, d);
    int cph = ctot / heads;
    int hh  = t / cph;
    float alpha = ex[(long)e * heads + hh] / (den[d * heads + hh] + 1e-16f);
    if (alpha_out != nullptr && t == 0) alpha_out[e] = alpha;
    atomicAdd(&out[(long)d * ctot + t], (float)feat[(long)s * ctot + t] * alpha);
}

// ---------------- bias + ELU in place (f32), optional f16 mirror for next GEMM ----------------
__global__ void gat_bias_elu(float* __restrict__ p, half_t* __restrict__ ph,
                             const float* __restrict__ bias, long n, int chmask)
{
    long i = (long)blockIdx.x * blockDim.x + threadIdx.x;
    if (i >= n) return;
    float v = p[i] + bias[(int)(i & chmask)];
    v = v > 0.f ? v : (expf(v) - 1.f);
    p[i] = v;
    if (ph != nullptr) ph[i] = (half_t)v;
}

// ---------------- global mean pool: sums/counts ----------------
__global__ void gat_pool_add(const float* __restrict__ h2a, const int* __restrict__ batch,
                             float* __restrict__ sums, float* __restrict__ cnt)
{
    long i = (long)blockIdx.x * blockDim.x + threadIdx.x;
    if (i >= (long)NN * CC) return;
    int nidx = (int)(i >> 6);
    int c    = (int)(i & 63);
    int g    = batch[nidx];
    atomicAdd(&sums[g * CC + c], h2a[i]);
    if (c == 0) atomicAdd(&cnt[g], 1.0f);
}

// ---------------- classifier head + log_softmax (one thread per graph) ----------------
__global__ void gat_final(const float* __restrict__ sums, const float* __restrict__ cnt,
                          const float* __restrict__ lin_w, const float* __restrict__ lin_b,
                          float* __restrict__ out)
{
    int g = blockIdx.x * blockDim.x + threadIdx.x;
    if (g >= NG) return;
    float inv = 1.0f / fmaxf(cnt[g], 1.0f);
    float lo[NCLS];
#pragma unroll
    for (int j = 0; j < NCLS; ++j) {
        float acc = 0.f;
        for (int i = 0; i < CC; ++i)
            acc += sums[g * CC + i] * lin_w[i * NCLS + j];
        lo[j] = acc * inv + lin_b[j];
    }
    float m = lo[0];
#pragma unroll
    for (int j = 1; j < NCLS; ++j) m = fmaxf(m, lo[j]);
    float s = 0.f;
#pragma unroll
    for (int j = 0; j < NCLS; ++j) s += expf(lo[j] - m);
    float ls = logf(s);
#pragma unroll
    for (int j = 0; j < NCLS; ++j) out[g * NCLS + j] = lo[j] - m - ls;
}

// ---------------- orchestration ----------------
extern "C" void kernel_launch(void* const* d_in, const int* in_sizes, int n_in,
                              void* d_out, int out_size, void* d_ws, size_t ws_size,
                              hipStream_t stream)
{
    (void)in_sizes; (void)n_in; (void)out_size; (void)ws_size;

    const float* x      = (const float*)d_in[0];
    const int*   ei     = (const int*)  d_in[1];
    const int*   batch  = (const int*)  d_in[2];
    const float* W1     = (const float*)d_in[3];
    const float* att1s  = (const float*)d_in[4];
    const float* att1d  = (const float*)d_in[5];
    const float* b1     = (const float*)d_in[6];
    const float* W2     = (const float*)d_in[7];
    const float* att2s  = (const float*)d_in[8];
    const float* att2d  = (const float*)d_in[9];
    const float* b2     = (const float*)d_in[10];
    const float* linw   = (const float*)d_in[11];
    const float* linb   = (const float*)d_in[12];

    float* out_logits = (float*)d_out;                 // NG*NCLS
    float* out_alpha  = (float*)d_out + NG * NCLS;     // ETOT

    // workspace layout (carved in float units; all chunk sizes multiple of 4 floats -> 16B aligned)
    float* ws = (float*)d_ws;
    half_t* xh   = (half_t*)ws;       ws += (long)NN * FIN / 2;   // f16 x
    half_t* W1p  = (half_t*)ws;       ws += (long)FIN * HC / 2;   // packed f16 W1
    half_t* W2p  = (half_t*)ws;       ws += (long)HC * CC / 2;    // packed f16 W2
    half_t* h1h  = (half_t*)ws;       ws += (long)NN * HC / 2;    // f16 h1 = x@W1
    float*  out1 = ws;                ws += (long)NN * HC;        // f32 aggregated layer1
    half_t* o1h  = (half_t*)ws;       ws += (long)NN * HC / 2;    // f16 elu(out1) for GEMM2
    half_t* h2h  = (half_t*)ws;       ws += (long)NN * CC / 2;    // f16 h2 = o1h@W2
    float*  out2 = ws;                ws += (long)NN * CC;        // f32 aggregated layer2
    float*  as1  = ws;                ws += (long)NN * HH1;
    float*  ad1  = ws;                ws += (long)NN * HH1;
    float*  emax1= ws;                ws += (long)NN * HH1;
    float*  den1 = ws;                ws += (long)NN * HH1;
    float*  ex1  = ws;                ws += (long)ETOT * HH1;
    float*  as2  = ws;                ws += (long)NN;
    float*  ad2  = ws;                ws += (long)NN;
    float*  emax2= ws;                ws += (long)NN;
    float*  den2 = ws;                ws += (long)NN;
    float*  ex2  = ws;                ws += (long)ETOT;
    float*  sums = ws;                ws += (long)NG * CC;
    float*  cnt  = ws;                ws += (long)NG;

    const float NEG_INF = -3.402823466e38f;
    auto blocks = [](long n, int t) { return (int)((n + t - 1) / t); };

    // ---- one-time precision conversion / weight packing ----
    gat_f32_to_f16<<<blocks((long)NN*FIN,256),256,0,stream>>>(x, xh, (long)NN*FIN);
    gat_pack_b<<<blocks((long)FIN*HC/16,256),256,0,stream>>>(W1, W1p, FIN, HC);
    gat_pack_b<<<blocks((long)HC*CC/16,256),256,0,stream>>>(W2, W2p, HC, CC);

    // ---- init accumulators ----
    gat_fill<<<blocks((long)NN*HH1,256),256,0,stream>>>(emax1, (long)NN*HH1, NEG_INF);
    gat_fill<<<blocks((long)NN*HH1,256),256,0,stream>>>(den1,  (long)NN*HH1, 0.f);
    gat_fill<<<blocks((long)NN*HC,256),256,0,stream>>>(out1,  (long)NN*HC,  0.f);
    gat_fill<<<blocks((long)NN,256),256,0,stream>>>(emax2, (long)NN, NEG_INF);
    gat_fill<<<blocks((long)NN,256),256,0,stream>>>(den2,  (long)NN, 0.f);
    gat_fill<<<blocks((long)NN*CC,256),256,0,stream>>>(out2,  (long)NN*CC,  0.f);
    gat_fill<<<blocks((long)NG*CC + NG,256),256,0,stream>>>(sums, (long)NG*CC + NG, 0.f);

    // ---- layer 1: projection (WMMA, 16x64 per wave), attention, segment-softmax, aggregate ----
    {
        int waves = (NN / 16) * (HC / 64);                // 25000
        gat_wmma_gemm<<<blocks(waves, 8), 256, 0, stream>>>(xh, W1p, h1h, NN, FIN, HC);
    }
    gat_attn_logits<<<blocks(NN, 8), 256, 0, stream>>>(h1h, att1s, att1d, as1, ad1, NN, HH1, CC);
    gat_edge_max<<<blocks((long)ETOT*HH1,256),256,0,stream>>>(ei, as1, ad1, emax1, HH1);
    gat_edge_exp<<<blocks((long)ETOT*HH1,256),256,0,stream>>>(ei, as1, ad1, emax1, ex1, den1, HH1);
    gat_edge_aggregate<<<ETOT, HC, 0, stream>>>(ei, h1h, ex1, den1, out1, nullptr, HH1, HC);
    gat_bias_elu<<<blocks((long)NN*HC,256),256,0,stream>>>(out1, o1h, b1, (long)NN*HC, HC - 1);

    // ---- layer 2 ----
    {
        int waves = (NN / 16) * (CC / 64);                // 3125
        gat_wmma_gemm<<<blocks(waves, 8), 256, 0, stream>>>(o1h, W2p, h2h, NN, HC, CC);
    }
    gat_attn_logits<<<blocks(NN, 8), 256, 0, stream>>>(h2h, att2s, att2d, as2, ad2, NN, 1, CC);
    gat_edge_max<<<blocks((long)ETOT,256),256,0,stream>>>(ei, as2, ad2, emax2, 1);
    gat_edge_exp<<<blocks((long)ETOT,256),256,0,stream>>>(ei, as2, ad2, emax2, ex2, den2, 1);
    gat_edge_aggregate<<<ETOT, CC, 0, stream>>>(ei, h2h, ex2, den2, out2, out_alpha, 1, CC);
    gat_bias_elu<<<blocks((long)NN*CC,256),256,0,stream>>>(out2, nullptr, b2, (long)NN*CC, CC - 1);

    // ---- pool + head ----
    gat_pool_add<<<blocks((long)NN*CC,256),256,0,stream>>>(out2, batch, sums, cnt);
    gat_final<<<1, NG, 0, stream>>>(sums, cnt, linw, linb, out_logits);
}
